// MultiheadAttentionWithBias_25065429139809
// MI455X (gfx1250) — matrix operand order repaired and checked by reading, not compiled, not run on hardware
//
#include <hip/hip_runtime.h>

typedef __attribute__((ext_vector_type(16))) _Float16 v16h;
typedef __attribute__((ext_vector_type(8)))  _Float16 v8h;
typedef __attribute__((ext_vector_type(4)))  _Float16 v4h;
typedef __attribute__((ext_vector_type(8)))  float    v8f;
typedef __attribute__((ext_vector_type(4)))  float    v4f;

#define BB   4
#define TT   1024
#define CC   768
#define HH   12
#define DD   64
#define NQKV (3 * CC)   // 2304

// ---------- fragment loaders (CDNA5 WMMA 16x16x32 f16 layouts) ----------
// A fragment 16x32 (MxK) from row-major source (K contiguous), leading dim ld.
// Per lane: two contiguous 8-half chunks -> two 16B loads.
__device__ __forceinline__ v16h load_a_f16(const _Float16* __restrict__ base,
                                           int ld, int row, int hi) {
  const _Float16* p = base + (size_t)row * ld + hi * 16;
  union { v16h v; v8h h[2]; } a;
  a.h[0] = *(const v8h*)(p);        // K = hi*16 + 0..7
  a.h[1] = *(const v8h*)(p + 8);    // K = hi*16 + 8..15
  return a.v;
}

// B fragment 32x16 (KxN) from a K-major source (K contiguous per column),
// leading dim ldk (elements between consecutive N columns).
// Per lane: 16 contiguous halves -> one 32B load.
__device__ __forceinline__ v16h load_b_f16_kmajor(const _Float16* __restrict__ base,
                                                  int ldk, int col, int hi) {
  const _Float16* p = base + (size_t)col * ldk + hi * 16;  // K = hi*16 + 0..15
  return *(const v16h*)p;
}

// ---------- f32 -> f16 conversion (vectorized) ----------
__global__ void __launch_bounds__(256) f32_to_f16_kernel(const float* __restrict__ src,
                                                         _Float16* __restrict__ dst, int n4) {
  int i = blockIdx.x * blockDim.x + threadIdx.x;
  int stride = gridDim.x * blockDim.x;
  for (; i < n4; i += stride) {
    v4f f = ((const v4f*)src)[i];
    v4h h;
    h[0] = (_Float16)f[0]; h[1] = (_Float16)f[1];
    h[2] = (_Float16)f[2]; h[3] = (_Float16)f[3];
    ((v4h*)dst)[i] = h;
  }
}

// ---------- f32 -> f16 transpose: dst[n][k] = src[k][n] ----------
__global__ void __launch_bounds__(256) transpose_f32_to_f16_kernel(
    const float* __restrict__ src, _Float16* __restrict__ dst, int rows, int cols) {
  __shared__ float tile[32][33];
  const int c0 = blockIdx.x * 32;
  const int r0 = blockIdx.y * 32;
  const int tx = threadIdx.x & 31;
  const int ty = threadIdx.x >> 5;     // 0..7
  #pragma unroll
  for (int i = ty; i < 32; i += 8)
    tile[i][tx] = src[(size_t)(r0 + i) * cols + c0 + tx];
  __syncthreads();
  #pragma unroll
  for (int i = ty; i < 32; i += 8)
    dst[(size_t)(c0 + i) * rows + r0 + tx] = (_Float16)tile[tx][i];
}

// ---------- QKV GEMM: X[4096x768] @ Wt^T + b -> Q,K [B,H,T,D], V [B,H,D,T] ----------
__global__ void __launch_bounds__(128) qkv_gemm_kernel(
    const _Float16* __restrict__ Xh, const _Float16* __restrict__ Wt,  // Wt: [2304][768]
    const float* __restrict__ bqkv,
    _Float16* __restrict__ Qh, _Float16* __restrict__ Kh, _Float16* __restrict__ Vt) {
  const int tid  = threadIdx.x;
  const int wave = tid >> 5;
  const int lane = tid & 31;
  const int row  = lane & 15;
  const int hi   = lane >> 4;

  const int m0 = blockIdx.x * 16;                    // 0..4095
  const int n0 = (blockIdx.y * 4 + wave) * 16;       // 0..2303

  v8f c = {};
  for (int k0 = 0; k0 < CC; k0 += 32) {
    v16h a = load_a_f16(Xh + (size_t)m0 * CC + k0, CC, row, hi);
    v16h b = load_b_f16_kmajor(Wt + (size_t)n0 * CC + k0, CC, row, hi);
    c = __builtin_amdgcn_wmma_f32_16x16x32_f16(false, a, false, b, (short)0, c, false, false);
  }

  const int col   = lane & 15;
  const int which = n0 / CC;            // 0=q 1=k 2=v (uniform within tile)
  const int h     = (n0 % CC) / DD;     // uniform within tile
  const int dbase = n0 % DD;            // {0,16,32,48}
  const int bidx  = m0 / TT;
  const int tb    = m0 % TT;
  const float badd = bqkv[n0 + col];

  if (which == 2) {
    // V transposed: Vt[b,h,d,t]; per lane t = tb+hi*8+g contiguous -> one 16B store
    union { v8h v; _Float16 e[8]; } st;
    #pragma unroll
    for (int g = 0; g < 8; ++g) st.e[g] = (_Float16)(c[g] + badd);
    _Float16* vp = Vt + (((size_t)bidx * HH + h) * DD + dbase + col) * TT + tb + hi * 8;
    *(v8h*)vp = st.v;
  } else {
    _Float16* dst = (which == 0) ? Qh : Kh;           // [B,H,T,D]
    #pragma unroll
    for (int g = 0; g < 8; ++g) {
      int t = tb + hi * 8 + g;
      dst[(((size_t)bidx * HH + h) * TT + t) * DD + dbase + col] = (_Float16)(c[g] + badd);
    }
  }
}

// ---------- Attention: logits + bias + mask, softmax, attn writeout, P@V ----------
__global__ void __launch_bounds__(128) attn_kernel(
    const _Float16* __restrict__ Qh, const _Float16* __restrict__ Kh,
    const _Float16* __restrict__ Vt,                    // [B,H,D,T] (t contiguous)
    const float* __restrict__ bias, const float* __restrict__ maskl,
    float* __restrict__ attn_out, _Float16* __restrict__ Yh) {
  __shared__ float S[16][TT];   // 64 KB: one 16-row stripe of logits/probs

  const int tid  = threadIdx.x;
  const int wave = tid >> 5;
  const int lane = tid & 31;
  const int row  = lane & 15;
  const int hi   = lane >> 4;

  const int bh = blockIdx.x;            // 0..47
  const int b  = bh / HH;
  const int h  = bh % HH;
  const int t0 = blockIdx.y * 16;

  const float scale = 0.125f;           // 1/sqrt(64)

  // Q fragments (K-dim = D = 64 -> two 32-wide steps)
  const _Float16* Qbase = Qh + ((size_t)bh * TT + t0) * DD;
  v16h aq0 = load_a_f16(Qbase +  0, DD, row, hi);
  v16h aq1 = load_a_f16(Qbase + 32, DD, row, hi);

  // ---- logits: each wave covers 16 of the 64 column tiles ----
  for (int st = wave; st < TT / 16; st += 4) {
    const int s0 = st * 16;
    // B = K^T: per lane one K row, d contiguous -> two 32B loads
    const _Float16* kp = Kh + ((size_t)bh * TT + s0 + row) * DD + hi * 16;
    v16h bk0 = *(const v16h*)(kp);        // d = hi*16 + 0..15
    v16h bk1 = *(const v16h*)(kp + 32);   // d = 32 + hi*16 + 0..15

    // prefetch next tile's bias/mask stream (dominant HBM traffic)
    if (st + 4 < TT / 16) {
      __builtin_prefetch(&bias[((size_t)bh * TT + t0 + row) * TT + s0 + 64], 0, 1);
      __builtin_prefetch(&maskl[((size_t)b * TT + t0 + row) * TT + s0 + 64], 0, 1);
    }

    v8f c = {};
    c = __builtin_amdgcn_wmma_f32_16x16x32_f16(false, aq0, false, bk0, (short)0, c, false, false);
    c = __builtin_amdgcn_wmma_f32_16x16x32_f16(false, aq1, false, bk1, (short)0, c, false, false);
    #pragma unroll
    for (int g = 0; g < 8; ++g) {
      int r = hi * 8 + g;
      int t = t0 + r;
      int s = s0 + row;
      S[r][s] = c[g] * scale
              + bias[((size_t)bh * TT + t) * TT + s]
              + maskl[((size_t)b * TT + t) * TT + s];
    }
  }
  __syncthreads();

  // ---- softmax: 8 lanes per row, intra-wave shfl reductions ----
  {
    const int r  = tid >> 3;
    const int g8 = tid & 7;
    float m = -3.0e38f;
    for (int j = g8; j < TT; j += 8) m = fmaxf(m, S[r][j]);
    m = fmaxf(m, __shfl_xor(m, 1, 32));
    m = fmaxf(m, __shfl_xor(m, 2, 32));
    m = fmaxf(m, __shfl_xor(m, 4, 32));
    float sum = 0.0f;
    for (int j = g8; j < TT; j += 8) {
      float e = __expf(S[r][j] - m);
      S[r][j] = e;
      sum += e;
    }
    sum += __shfl_xor(sum, 1, 32);
    sum += __shfl_xor(sum, 2, 32);
    sum += __shfl_xor(sum, 4, 32);
    float inv = 1.0f / sum;
    for (int j = g8; j < TT; j += 8) S[r][j] *= inv;
  }
  __syncthreads();

  // ---- write normalized attn (float4, coalesced b128 stores) ----
  {
    float* arow = attn_out + ((size_t)bh * TT + t0) * TT;
    for (int idx = tid * 4; idx < 16 * TT; idx += 128 * 4) {
      int r = idx >> 10;
      int s = idx & (TT - 1);
      *(v4f*)&arow[(size_t)r * TT + s] = *(const v4f*)&S[r][s];
    }
  }

  // ---- y tile = P[16x1024] @ V[1024x64]; wave -> one 16-wide N tile ----
  {
    const int n0 = wave * 16;   // D = 64 = 4 waves * 16
    v8f c = {};
    for (int k0 = 0; k0 < TT; k0 += 32) {
      // A from LDS probs: four ds_load_b128 + pack-convert
      union { v16h v; _Float16 e[16]; } a;
      {
        const float* p = &S[row][k0 + hi * 16];
        v4f f0 = *(const v4f*)(p);
        v4f f1 = *(const v4f*)(p + 4);
        v4f f2 = *(const v4f*)(p + 8);
        v4f f3 = *(const v4f*)(p + 12);
        #pragma unroll
        for (int i = 0; i < 4; ++i) {
          a.e[i]      = (_Float16)f0[i];
          a.e[4 + i]  = (_Float16)f1[i];
          a.e[8 + i]  = (_Float16)f2[i];
          a.e[12 + i] = (_Float16)f3[i];
        }
      }
      // B from Vt [d][t]: per lane fixed d, s contiguous -> one 32B load
      v16h bv = load_b_f16_kmajor(Vt + ((size_t)bh * DD + n0) * TT + k0, TT, row, hi);
      c = __builtin_amdgcn_wmma_f32_16x16x32_f16(false, a.v, false, bv, (short)0, c, false, false);
    }
    #pragma unroll
    for (int g = 0; g < 8; ++g) {
      int t = t0 + hi * 8 + g;
      Yh[((size_t)b * TT + t) * CC + h * DD + n0 + row] = (_Float16)c[g];
    }
  }
}

// ---------- Output projection: Y[4096x768] @ Wt^T + b -> f32 out ----------
__global__ void __launch_bounds__(128) proj_gemm_kernel(
    const _Float16* __restrict__ Yh, const _Float16* __restrict__ Wt,  // Wt: [768][768]
    const float* __restrict__ bproj, float* __restrict__ out) {
  const int tid  = threadIdx.x;
  const int wave = tid >> 5;
  const int lane = tid & 31;
  const int row  = lane & 15;
  const int hi   = lane >> 4;

  const int m0 = blockIdx.x * 16;
  const int n0 = (blockIdx.y * 4 + wave) * 16;

  v8f c = {};
  for (int k0 = 0; k0 < CC; k0 += 32) {
    v16h a = load_a_f16(Yh + (size_t)m0 * CC + k0, CC, row, hi);
    v16h b = load_b_f16_kmajor(Wt + (size_t)n0 * CC + k0, CC, row, hi);
    c = __builtin_amdgcn_wmma_f32_16x16x32_f16(false, a, false, b, (short)0, c, false, false);
  }
  const int col = lane & 15;
  const float badd = bproj[n0 + col];
  #pragma unroll
  for (int g = 0; g < 8; ++g) {
    int m = m0 + hi * 8 + g;
    out[(size_t)m * CC + n0 + col] = c[g] + badd;
  }
}

extern "C" void kernel_launch(void* const* d_in, const int* in_sizes, int n_in,
                              void* d_out, int out_size, void* d_ws, size_t ws_size,
                              hipStream_t stream) {
  (void)in_sizes; (void)n_in; (void)out_size; (void)ws_size;

  const float* x     = (const float*)d_in[0];
  const float* maskl = (const float*)d_in[1];
  const float* bias  = (const float*)d_in[2];
  const float* Wqkv  = (const float*)d_in[3];
  const float* bqkv  = (const float*)d_in[4];
  const float* Wproj = (const float*)d_in[5];
  const float* bproj = (const float*)d_in[6];

  char* ws = (char*)d_ws;                             // ~34.5 MB used
  _Float16* Xh  = (_Float16*)(ws);                    //  6,291,456 B  [4096][768]
  _Float16* Wqt = (_Float16*)(ws +  6291456);         //  3,538,944 B  [2304][768]
  _Float16* Wpt = (_Float16*)(ws +  9830400);         //  1,179,648 B  [768][768]
  _Float16* Qh  = (_Float16*)(ws + 11010048);         //  6,291,456 B  [B,H,T,D]
  _Float16* Kh  = (_Float16*)(ws + 17301504);         //  6,291,456 B  [B,H,T,D]
  _Float16* Vt  = (_Float16*)(ws + 23592960);         //  6,291,456 B  [B,H,D,T]
  _Float16* Yh  = (_Float16*)(ws + 29884416);         //  6,291,456 B  [4096][768]

  float* y_out    = (float*)d_out;                    // [B,T,C]
  float* attn_out = y_out + (size_t)BB * TT * CC;     // [B,H,T,T]

  f32_to_f16_kernel<<<1024, 256, 0, stream>>>(x, Xh, BB * TT * CC / 4);
  transpose_f32_to_f16_kernel<<<dim3(NQKV / 32, CC / 32), 256, 0, stream>>>(Wqkv, Wqt, CC, NQKV);
  transpose_f32_to_f16_kernel<<<dim3(CC / 32, CC / 32), 256, 0, stream>>>(Wproj, Wpt, CC, CC);

  qkv_gemm_kernel<<<dim3(256, 36), 128, 0, stream>>>(Xh, Wqt, bqkv, Qh, Kh, Vt);
  attn_kernel<<<dim3(BB * HH, TT / 16), 128, 0, stream>>>(Qh, Kh, Vt, bias, maskl,
                                                          attn_out, Yh);
  proj_gemm_kernel<<<dim3(256, 12), 128, 0, stream>>>(Yh, Wpt, bproj, y_out);
}
